// DecoderLayer_24172075942013
// MI455X (gfx1250) — matrix operand order repaired and checked by reading, not compiled, not run on hardware
//
#include <hip/hip_runtime.h>
#include <hip/hip_bf16.h>

typedef __attribute__((ext_vector_type(16))) _Float16 v16h;
typedef __attribute__((ext_vector_type(8)))  float    v8f;

#define D_MODEL 512
#define HEADS   8
#define DK      64
#define D_FF    1024
#define B_      4
#define S_      2048
#define NTOK    (B_ * S_)          // 8192 tokens

#define LOG2E   1.44269504088896340736f

union AFrag { v16h v; unsigned u[8]; _Float16 h[16]; };
union CFrag { v8f  v; float f[8]; };

// ---------------------------------------------------------------------------
// WMMA fragment loaders (CDNA5 16x16x32 f16 layouts, wave32)
// ---------------------------------------------------------------------------

__device__ inline v8f wmma_f16(v16h a, v16h b, v8f c) {
    return __builtin_amdgcn_wmma_f32_16x16x32_f16(false, a, false, b,
                                                  (short)0, c, false, false);
}

// A-matrix 16x32: element (m,k) at src[m*ld + k].  Lane l<16: row l, K 0-7 &
// 16-23; lane>=16: row l-16, K 8-15 & 24-31 (2 halves per VGPR).
__device__ inline v16h load_a_frag(const _Float16* src, int ld) {
    AFrag a;
    const int lane = threadIdx.x & 31;
    const int m    = lane & 15;
    const int kb   = (lane >> 4) * 8;
    const _Float16* row = src + m * ld;
#pragma unroll
    for (int j = 0; j < 8; ++j) {
        const int k = ((j & 4) << 2) + kb + 2 * (j & 3);
        a.u[j] = *(const unsigned*)(row + k);
    }
    return a.v;
}

// B-matrix 32x16 from TRANSPOSED (n-major) storage: element (k,n) at
// src[n*ld + k].  Contiguous pairs -> compiler merges into ds_load_b128.
__device__ inline v16h load_b_frag_T(const _Float16* src, int ld) {
    AFrag b;
    const int lane = threadIdx.x & 31;
    const int n    = lane & 15;
    const int kb   = (lane >> 4) * 8;
    const _Float16* col = src + n * ld;
#pragma unroll
    for (int j = 0; j < 8; ++j)
        b.u[j] = *(const unsigned*)(col + ((j & 4) << 2) + kb + 2 * (j & 3));
    return b.v;
}

// ---------------------------------------------------------------------------
// Small utility kernels
// ---------------------------------------------------------------------------

__global__ void copy4_kernel(const float4* __restrict__ in,
                             float4* __restrict__ out, int n4) {
    int i = blockIdx.x * blockDim.x + threadIdx.x;
    if (i < n4) out[i] = in[i];
}

__global__ void cvt_f16_kernel(const float* __restrict__ in,
                               _Float16* __restrict__ out, int n4) {
    int i = blockIdx.x * blockDim.x + threadIdx.x;
    if (i < n4) {
        float4 v = ((const float4*)in)[i];
        union { _Float16 h[4]; unsigned long long u; } p;
        p.h[0] = (_Float16)v.x; p.h[1] = (_Float16)v.y;
        p.h[2] = (_Float16)v.z; p.h[3] = (_Float16)v.w;
        ((unsigned long long*)out)[i] = p.u;
    }
}

// mask [B,1,S,S] int32 -> bit-packed words, word w covers keys [w*32, w*32+32)
__global__ void pack_mask_kernel(const int* __restrict__ mask,
                                 unsigned* __restrict__ bits, int nwords) {
    int w = blockIdx.x * blockDim.x + threadIdx.x;
    if (w >= nwords) return;
    const int* p = mask + (size_t)w * 32;
    unsigned acc = 0;
#pragma unroll
    for (int i = 0; i < 32; ++i) acc |= (p[i] != 0) ? (1u << i) : 0u;
    bits[w] = acc;
}

// W [K][N] fp32 -> WT [N][K] f16 (64x64 LDS-tiled transpose, one-time cost)
__global__ __launch_bounds__(256)
void transpose_w_kernel(const float* __restrict__ W, _Float16* __restrict__ WT,
                        int K, int N) {
    __shared__ _Float16 T[64][72];
    const int k0 = blockIdx.y * 64, n0 = blockIdx.x * 64;
    const int tid = threadIdx.x;
#pragma unroll
    for (int c = 0; c < 4; ++c) {                 // 64x64 fp32 in, 1024 float4
        const int idx = tid + c * 256;
        const int r = idx >> 4, col = (idx & 15) * 4;
        float4 v = *(const float4*)(W + (size_t)(k0 + r) * N + n0 + col);
        union { _Float16 h[4]; unsigned long long u; } p;
        p.h[0] = (_Float16)v.x; p.h[1] = (_Float16)v.y;
        p.h[2] = (_Float16)v.z; p.h[3] = (_Float16)v.w;
        *(unsigned long long*)(&T[r][col]) = p.u;
    }
    __syncthreads();
#pragma unroll
    for (int c = 0; c < 2; ++c) {                 // out rows n, contiguous k
        const int idx = tid + c * 256;
        const int r = idx >> 3, col = (idx & 7) * 8;
        union { _Float16 h[8]; uint4 u; } p;
#pragma unroll
        for (int i = 0; i < 8; ++i) p.h[i] = T[col + i][r];
        *(uint4*)(WT + (size_t)(n0 + r) * K + k0 + col) = p.u;
    }
}

// act [B,S,512] f16 -> actT [B,H,64,S] f16 (dk-major per head)
__global__ __launch_bounds__(256)
void transpose_act_kernel(const _Float16* __restrict__ in,
                          _Float16* __restrict__ outT) {
    __shared__ _Float16 T[64][72];
    const int s0 = blockIdx.x * 64;
    const int h  = blockIdx.y;
    const int b  = blockIdx.z;
    const int tid = threadIdx.x;
#pragma unroll
    for (int c = 0; c < 2; ++c) {                 // 64 s-rows x 64 dk
        const int idx = tid + c * 256;
        const int r = idx >> 3, col = (idx & 7) * 8;
        *(uint4*)(&T[r][col]) = *(const uint4*)(
            in + (size_t)(b * S_ + s0 + r) * D_MODEL + h * DK + col);
    }
    __syncthreads();
#pragma unroll
    for (int c = 0; c < 2; ++c) {                 // out rows d, contiguous s
        const int idx = tid + c * 256;
        const int r = idx >> 3, col = (idx & 7) * 8;
        union { _Float16 h[8]; uint4 u; } p;
#pragma unroll
        for (int i = 0; i < 8; ++i) p.h[i] = T[col + i][r];
        *(uint4*)(outT + (size_t)((b * HEADS + h) * DK + r) * S_ + s0 + col)
            = p.u;
    }
}

// LayerNorm over last dim (512): one wave per token, f16 output.
__global__ __launch_bounds__(256)
void ln_kernel(const float* __restrict__ x, const float* __restrict__ g,
               const float* __restrict__ be, _Float16* __restrict__ out) {
    const int tok  = (blockIdx.x * blockDim.x + threadIdx.x) >> 5;
    const int lane = threadIdx.x & 31;
    const float* row = x + (size_t)tok * D_MODEL;
    float v[16];
    const float4* r4 = (const float4*)(row + lane * 16);
#pragma unroll
    for (int i = 0; i < 4; ++i) {
        float4 t = r4[i];
        v[4*i] = t.x; v[4*i+1] = t.y; v[4*i+2] = t.z; v[4*i+3] = t.w;
    }
    float s = 0.f, s2 = 0.f;
#pragma unroll
    for (int i = 0; i < 16; ++i) { s += v[i]; s2 += v[i] * v[i]; }
#pragma unroll
    for (int msk = 1; msk < 32; msk <<= 1) {
        s  += __shfl_xor(s,  msk, 32);
        s2 += __shfl_xor(s2, msk, 32);
    }
    const float mu = s * (1.f / D_MODEL);
    const float rs = rsqrtf(s2 * (1.f / D_MODEL) - mu * mu + 1e-5f);
    union { _Float16 h[16]; uint4 u4[2]; } o;
#pragma unroll
    for (int i = 0; i < 16; ++i) {
        const int c = lane * 16 + i;
        o.h[i] = (_Float16)((v[i] - mu) * rs * g[c] + be[c]);
    }
    uint4* dst = (uint4*)(out + (size_t)tok * D_MODEL + lane * 16);
    dst[0] = o.u4[0]; dst[1] = o.u4[1];
}

// ---------------------------------------------------------------------------
// Flash attention: block = 4 waves, 64 query rows for one (b,h).
// kvsrc : row-major [B,S,512]   -> K side (Q.K^T, contiguous-dk B reads)
// kvT   : dk-major  [B,H,64,S]  -> V side (P.V,  contiguous-key B reads)
// Software-pipelined: next step's 64-key tiles are fetched into registers
// (clause'd global_load_b128) while the current step computes; one barrier
// per step with double-buffered LDS.  One online-softmax pass per 64 keys
// in the exp2 domain (scale*log2(e) folded into Q fragments).
// ---------------------------------------------------------------------------
template <bool MASKED>
__global__ __launch_bounds__(128)
void flash_attn_kernel(const _Float16* __restrict__ qsrc,
                       const _Float16* __restrict__ kvsrc,
                       const _Float16* __restrict__ kvT,
                       const unsigned* __restrict__ maskbits,
                       float* __restrict__ xout) {
    __shared__ _Float16 Kt [2][64][72];    // 64 keys x 64 dk (+8 pad), x2
    __shared__ _Float16 VTs[2][64][72];    // 64 dk x 64 keys (+8 pad), x2
    __shared__ _Float16 Pt[4][16][66];     // per-wave 16x64 P buffer (+2 pad)

    const int tid  = threadIdx.x;
    const int lane = tid & 31;
    const int w    = tid >> 5;
    const int h    = blockIdx.y;
    const int b    = blockIdx.z;
    const int bh   = b * HEADS + h;
    const int q0   = blockIdx.x * 64 + w * 16;   // first query row for wave

    // per-thread fill coordinates (8-half chunks, 4 per tile per thread)
    const int fr = tid >> 3;                 // row 0..15 (+32 per chunk pair)
    const int fc = (tid & 7) * 8;            // col 0..56

    // Q fragments (two 16x32 dk-chunks); fold 1/sqrt(dk)*log2(e) into them.
    const int m   = lane & 15;
    const int kb8 = (lane >> 4) * 8;
    v16h qa0, qa1;
    {
        const _Float16* qrow =
            qsrc + ((size_t)(b * S_ + q0 + m) * D_MODEL + h * DK);
        AFrag a0, a1;
#pragma unroll
        for (int j = 0; j < 8; ++j) {
            const int k = ((j & 4) << 2) + kb8 + 2 * (j & 3);
            a0.u[j] = *(const unsigned*)(qrow + k);
            a1.u[j] = *(const unsigned*)(qrow + 32 + k);
        }
        const _Float16 qs = (_Float16)(0.125f * LOG2E);
#pragma unroll
        for (int i = 0; i < 16; ++i) { a0.h[i] *= qs; a1.h[i] *= qs; }
        qa0 = a0.v; qa1 = a1.v;
    }

    v8f O[4] = {};                 // 16 q-rows x 64 dk, fp32
    float mrow[8], lrow[8];
#pragma unroll
    for (int j = 0; j < 8; ++j) { mrow[j] = -3e38f; lrow[j] = 0.f; }

    const float MASKVAL = -1e9f * LOG2E;   // exact jnp.where(-1e9), log2 dom
    const int   hi      = lane >> 4;
    const int   n       = lane & 15;

    // prefetch step-0 tiles into registers
    uint4 kreg[4], vreg[4];
#pragma unroll
    for (int c = 0; c < 4; ++c) {
        kreg[c] = *(const uint4*)(
            kvsrc + (size_t)(b * S_ + fr + c * 16) * D_MODEL + h * DK + fc);
        vreg[c] = *(const uint4*)(
            kvT + (size_t)(bh * DK + fr + c * 16) * S_ + fc);
    }

    for (int step = 0; step < S_ / 64; ++step) {
        const int buf = step & 1;
        // commit prefetched tiles to LDS buffer `buf`
#pragma unroll
        for (int c = 0; c < 4; ++c) {
            *(uint4*)(&Kt [buf][fr + c * 16][fc]) = kreg[c];
            *(uint4*)(&VTs[buf][fr + c * 16][fc]) = vreg[c];
        }
        // issue next step's global loads; in flight during compute below
        if (step + 1 < S_ / 64) {
            const int k1 = (step + 1) * 64;
#pragma unroll
            for (int c = 0; c < 4; ++c) {
                kreg[c] = *(const uint4*)(
                    kvsrc + (size_t)(b * S_ + k1 + fr + c * 16) * D_MODEL
                          + h * DK + fc);
                vreg[c] = *(const uint4*)(
                    kvT + (size_t)(bh * DK + fr + c * 16) * S_ + k1 + fc);
            }
        }
        __syncthreads();

        // scores S[16 q][64 keys] -> four 16x16 C tiles (8 WMMAs)
        CFrag sc[4];
#pragma unroll
        for (int t = 0; t < 4; ++t) {
            const v16h b0 = load_b_frag_T(&Kt[buf][t * 16][0],  72);
            const v16h b1 = load_b_frag_T(&Kt[buf][t * 16][32], 72);
            v8f acc = v8f{};
            acc = wmma_f16(qa0, b0, acc);
            acc = wmma_f16(qa1, b1, acc);
            sc[t].v = acc;
        }

        unsigned mw0[8], mw1[8];
        if (MASKED) {
#pragma unroll
            for (int j = 0; j < 8; ++j) {
                const size_t base =
                    (size_t)(b * S_ + q0 + j + 8 * hi) * (S_ / 32) + step * 2;
                mw0[j] = maskbits[base];
                mw1[j] = maskbits[base + 1];
            }
        }

        // single online-softmax pass over all 64 keys
#pragma unroll
        for (int j = 0; j < 8; ++j) {
            float a0 = sc[0].f[j], a1 = sc[1].f[j];
            float a2 = sc[2].f[j], a3 = sc[3].f[j];
            if (MASKED) {
                if (!((mw0[j] >> n)        & 1)) a0 = MASKVAL;
                if (!((mw0[j] >> (16 + n)) & 1)) a1 = MASKVAL;
                if (!((mw1[j] >> n)        & 1)) a2 = MASKVAL;
                if (!((mw1[j] >> (16 + n)) & 1)) a3 = MASKVAL;
            }
            float r = fmaxf(fmaxf(a0, a1), fmaxf(a2, a3));
            r = fmaxf(r, __shfl_xor(r, 1, 32));
            r = fmaxf(r, __shfl_xor(r, 2, 32));
            r = fmaxf(r, __shfl_xor(r, 4, 32));
            r = fmaxf(r, __shfl_xor(r, 8, 32));
            const float mn    = fmaxf(mrow[j], r);
            const float alpha = exp2f(mrow[j] - mn);
            const float p0 = exp2f(a0 - mn);
            const float p1 = exp2f(a1 - mn);
            const float p2 = exp2f(a2 - mn);
            const float p3 = exp2f(a3 - mn);
            const int   pr = j + 8 * hi;
            Pt[w][pr][n]      = (_Float16)p0;      // C-layout -> LDS
            Pt[w][pr][16 + n] = (_Float16)p1;
            Pt[w][pr][32 + n] = (_Float16)p2;
            Pt[w][pr][48 + n] = (_Float16)p3;
            float t2 = (p0 + p1) + (p2 + p3);
            t2 += __shfl_xor(t2, 1, 32);
            t2 += __shfl_xor(t2, 2, 32);
            t2 += __shfl_xor(t2, 4, 32);
            t2 += __shfl_xor(t2, 8, 32);
            lrow[j] = lrow[j] * alpha + t2;
            mrow[j] = mn;
#pragma unroll
            for (int nt = 0; nt < 4; ++nt) O[nt][j] *= alpha;
        }

        // O += P(16x64) . V(64x64)   (8 WMMAs, K split 2x32)
        const v16h pf0 = load_a_frag(&Pt[w][0][0],  66);
        const v16h pf1 = load_a_frag(&Pt[w][0][32], 66);
#pragma unroll
        for (int nt = 0; nt < 4; ++nt) {
            const v16h v0 = load_b_frag_T(&VTs[buf][nt * 16][0],  72);
            const v16h v1 = load_b_frag_T(&VTs[buf][nt * 16][32], 72);
            O[nt] = wmma_f16(pf0, v0, O[nt]);
            O[nt] = wmma_f16(pf1, v1, O[nt]);
        }
    }

    // epilogue: normalize rows, add residual into running x (disjoint cols)
#pragma unroll
    for (int j = 0; j < 8; ++j) {
        const float inv = 1.f / lrow[j];
        float* dst = xout + ((size_t)(b * S_ + q0 + j + 8 * hi) * D_MODEL
                             + h * DK + n);
#pragma unroll
        for (int nt = 0; nt < 4; ++nt) dst[nt * 16] += O[nt][j] * inv;
    }
}

// ---------------------------------------------------------------------------
// Tiled WMMA GEMM, B pre-transposed (BT [N][K] f16).
// Block = 4 waves -> 64x128 C tile, K-step 64 (16 WMMAs per barrier pair).
// MODE 0: out f16 = relu(A.B + bias)          (FFN layer 1)
// MODE 1: out f32 += A.B + bias               (FFN layer 2 + residual)
// ---------------------------------------------------------------------------
template <int MODE>
__global__ __launch_bounds__(128)
void gemm_kernel(const _Float16* __restrict__ A, const _Float16* __restrict__ BT,
                 const float* __restrict__ bias, void* __restrict__ outp,
                 int M, int N, int K) {
    __shared__ _Float16 As [64][72];    // 64 m x 64 k (+8 pad)
    __shared__ _Float16 BsT[128][72];   // 128 n x 64 k (+8 pad)
    const int tid  = threadIdx.x;
    const int lane = tid & 31;
    const int w    = tid >> 5;
    const int m0   = blockIdx.y * 64;
    const int n0   = blockIdx.x * 128;

    v8f acc[8] = {};
    for (int k0 = 0; k0 < K; k0 += 64) {
#pragma unroll
        for (int c = 0; c < 4; ++c) {            // As: 512 chunks of 8 halves
            const int idx = tid + c * 128;
            const int r = idx >> 3, col = (idx & 7) * 8;
            *(uint4*)(&As[r][col]) =
                *(const uint4*)(A + (size_t)(m0 + r) * K + k0 + col);
        }
#pragma unroll
        for (int c = 0; c < 8; ++c) {            // BsT: 1024 chunks
            const int idx = tid + c * 128;
            const int r = idx >> 3, col = (idx & 7) * 8;
            *(uint4*)(&BsT[r][col]) =
                *(const uint4*)(BT + (size_t)(n0 + r) * K + k0 + col);
        }
        if (k0 + 64 < K) {                       // prefetch next A/B tiles
            __builtin_prefetch(A + (size_t)(m0 + (tid >> 1)) * K + k0 + 64
                                 + (tid & 1) * 32, 0, 1);
            __builtin_prefetch(BT + (size_t)(n0 + tid) * K + k0 + 64, 0, 1);
        }
        __syncthreads();
        const v16h a0 = load_a_frag(&As[w * 16][0],  72);
        const v16h a1 = load_a_frag(&As[w * 16][32], 72);
#pragma unroll
        for (int nt = 0; nt < 8; ++nt) {
            const v16h b0 = load_b_frag_T(&BsT[nt * 16][0],  72);
            const v16h b1 = load_b_frag_T(&BsT[nt * 16][32], 72);
            acc[nt] = wmma_f16(a0, b0, acc[nt]);
            acc[nt] = wmma_f16(a1, b1, acc[nt]);
        }
        __syncthreads();
    }

    const int hi = lane >> 4, n = lane & 15;
#pragma unroll
    for (int j = 0; j < 8; ++j) {
        const int row = m0 + w * 16 + j + 8 * hi;
#pragma unroll
        for (int nt = 0; nt < 8; ++nt) {
            const int col = n0 + nt * 16 + n;
            const float v = acc[nt][j] + bias[col];
            if (MODE == 0) {
                ((_Float16*)outp)[(size_t)row * N + col] =
                    (_Float16)fmaxf(v, 0.f);
            } else {
                ((float*)outp)[(size_t)row * N + col] += v;
            }
        }
    }
}

// ---------------------------------------------------------------------------
// Host launch
// ---------------------------------------------------------------------------
extern "C" void kernel_launch(void* const* d_in, const int* in_sizes, int n_in,
                              void* d_out, int out_size, void* d_ws,
                              size_t ws_size, hipStream_t stream) {
    const float* x    = (const float*)d_in[0];
    const float* enc  = (const float*)d_in[1];
    const int*   mask = (const int*)d_in[2];
    const float* ln1g = (const float*)d_in[3];
    const float* ln1b = (const float*)d_in[4];
    const float* ln2g = (const float*)d_in[5];
    const float* ln2b = (const float*)d_in[6];
    const float* ln3g = (const float*)d_in[7];
    const float* ln3b = (const float*)d_in[8];
    const float* W1   = (const float*)d_in[9];
    const float* b1   = (const float*)d_in[10];
    const float* W2   = (const float*)d_in[11];
    const float* b2   = (const float*)d_in[12];
    float* out = (float*)d_out;

    char* ws = (char*)d_ws;
    _Float16* t16    = (_Float16*)(ws);                      //  8 MiB
    _Float16* enc16  = (_Float16*)(ws + (8u  << 20));        //  8 MiB
    _Float16* w1T    = (_Float16*)(ws + (16u << 20));        //  1 MiB
    _Float16* w2T    = (_Float16*)(ws + (17u << 20));        //  1 MiB
    _Float16* h16    = (_Float16*)(ws + (18u << 20));        // 16 MiB
    unsigned* mbits  = (unsigned*)(ws + (34u << 20));        //  2 MiB
    _Float16* vT16   = (_Float16*)(ws + (36u << 20));        //  8 MiB
    _Float16* encT16 = (_Float16*)(ws + (44u << 20));        //  8 MiB

    const int n_x4 = NTOK * D_MODEL / 4;                     // 1,048,576
    copy4_kernel<<<n_x4 / 256, 256, 0, stream>>>((const float4*)x,
                                                 (float4*)out, n_x4);
    cvt_f16_kernel<<<n_x4 / 256, 256, 0, stream>>>(enc, enc16, n_x4);
    transpose_w_kernel<<<dim3(D_FF / 64, D_MODEL / 64), 256, 0, stream>>>(
        W1, w1T, D_MODEL, D_FF);
    transpose_w_kernel<<<dim3(D_MODEL / 64, D_FF / 64), 256, 0, stream>>>(
        W2, w2T, D_FF, D_MODEL);
    const int nwords = B_ * S_ * (S_ / 32);                  // 524,288
    pack_mask_kernel<<<nwords / 256, 256, 0, stream>>>(mask, mbits, nwords);

    const dim3 lnG(NTOK * 32 / 256);                         // 1024 blocks
    const dim3 fg(S_ / 64, HEADS, B_);                       // 32 x 8 x 4
    const dim3 tg(S_ / 64, HEADS, B_);

    transpose_act_kernel<<<tg, 256, 0, stream>>>(enc16, encT16);

    // x += SelfAttn(LN1(x)) with mask
    ln_kernel<<<lnG, 256, 0, stream>>>(out, ln1g, ln1b, t16);
    transpose_act_kernel<<<tg, 256, 0, stream>>>(t16, vT16);
    flash_attn_kernel<true><<<fg, 128, 0, stream>>>(t16, t16, vT16, mbits,
                                                    out);

    // x += CrossAttn(LN2(x), enc)
    ln_kernel<<<lnG, 256, 0, stream>>>(out, ln2g, ln2b, t16);
    flash_attn_kernel<false><<<fg, 128, 0, stream>>>(t16, enc16, encT16,
                                                     nullptr, out);

    // x += FFN(LN3(x))
    ln_kernel<<<lnG, 256, 0, stream>>>(out, ln3g, ln3b, t16);
    const dim3 g1(D_FF / 128, NTOK / 64);                    // 8 x 128
    gemm_kernel<0><<<g1, 128, 0, stream>>>(t16, w1T, b1, h16,
                                           NTOK, D_FF, D_MODEL);
    const dim3 g2(D_MODEL / 128, NTOK / 64);                 // 4 x 128
    gemm_kernel<1><<<g2, 128, 0, stream>>>(h16, w2T, b2, out,
                                           NTOK, D_MODEL, D_FF);
}
